// XLSTM_dynamic_graph_31748398252686
// MI455X (gfx1250) — compile-verified, matrix operand order checked by reading, not run on hardware
//
#include <hip/hip_runtime.h>
#include <hip/hip_bf16.h>
#include <math.h>

typedef __attribute__((ext_vector_type(16))) __bf16 v16bf;
typedef __attribute__((ext_vector_type(8)))  float  v8f;

#define T_STEPS 33
#define BATCH   16
#define NNODES  512
#define DMODEL  128
#define LSEQ    96
#define CIN     3
#define KSCALE  0.08838834764831845f   /* 1/sqrt(128) */
#define EPS_F   1e-5f
#define TBN_TOT (T_STEPS * BATCH * NNODES)   /* 270336 */
#define BN_TOT  (BATCH * NNODES)             /* 8192 */

__device__ __forceinline__ float softplus_f(float x) {
  return (x > 20.f) ? x : log1pf(expf(x));
}

// Load a 16x32 bf16 fragment (A layout; also B=X^T when X is row-major [n][k])
// from a row-major [16+][ld] bf16 buffer, per CDNA5 ISA 16-bit A-matrix layout:
// lane half selects K-subblock; VGPR v<4 -> K = hi*8+2v, v>=4 -> K = 16+hi*8+2(v-4).
// (Lowers to two b128 loads per K-step fragment.)
__device__ __forceinline__ v16bf load_frag_rowmajor(const __bf16* base, int ld,
                                                    int kbase, int lane) {
  int r  = lane & 15;
  int hi = lane >> 4;
  const __bf16* row = base + (size_t)r * ld;
  v16bf f;
#pragma unroll
  for (int v = 0; v < 8; ++v) {
    int k = kbase + ((v < 4) ? (hi * 8 + 2 * v) : (16 + hi * 8 + 2 * (v - 4)));
    f[2 * v]     = row[k];
    f[2 * v + 1] = row[k + 1];
  }
  return f;
}

// Async copy of one 32KB chunk (128 nodes x 128 bf16) global -> LDS.
// 8 wave-wide GLOBAL_LOAD_ASYNC_TO_LDS_B128 issues per wave (tracked by ASYNCcnt).
__device__ __forceinline__ void async_copy_chunk(const __bf16* gsrc, uint32_t lds_base,
                                                 int tid) {
#pragma unroll
  for (int i = 0; i < 8; ++i) {
    int e = i * 256 + tid;                       // 16-byte unit index, 2048 total
    uint32_t l = lds_base + (uint32_t)e * 16u;
    uint64_t g = (uint64_t)(uintptr_t)gsrc + (uint64_t)e * 16u;
    asm volatile("global_load_async_to_lds_b128 %0, %1, off"
                 :: "v"(l), "v"(g) : "memory");
  }
}

__device__ __forceinline__ void wait_asynccnt_8() {
  asm volatile("s_wait_asynccnt 0x8" ::: "memory");
}
__device__ __forceinline__ void wait_asynccnt_0() {
  asm volatile("s_wait_asynccnt 0x0" ::: "memory");
}

// ---------------- kernel 0: weights fp32 -> bf16 ----------------
__global__ void k0_convert(const float* __restrict__ Wk, const float* __restrict__ Wv,
                           __bf16* __restrict__ Wkb, __bf16* __restrict__ Wvb) {
  int i = blockIdx.x * blockDim.x + threadIdx.x;
  if (i < DMODEL * DMODEL) {
    Wkb[i] = (__bf16)Wk[i];
    Wvb[i] = (__bf16)Wv[i];
  }
}

// ---------------- kernel 1: patchify + 1x1 conv + BN -> xt (bf16) ----------------
// xt layout: [(t*16+b)*512 + node][128]
__global__ void k1_patch_conv_bn(const float* __restrict__ x,
                                 const float* __restrict__ Wconv,
                                 const float* __restrict__ bconv,
                                 const float* __restrict__ bng,
                                 const float* __restrict__ bnb,
                                 const float* __restrict__ bnm,
                                 const float* __restrict__ bnv,
                                 __bf16* __restrict__ xt) {
  __shared__ float sWc[DMODEL * 9];
  __shared__ float sScale[DMODEL], sShift[DMODEL], sBc[DMODEL];
  int tid = threadIdx.x;
  for (int e = tid; e < DMODEL * 9; e += blockDim.x) sWc[e] = Wconv[e];
  for (int d = tid; d < DMODEL; d += blockDim.x) {
    float sc = bng[d] * rsqrtf(bnv[d] + EPS_F);
    sScale[d] = sc;
    sShift[d] = bnb[d] - bnm[d] * sc;
    sBc[d] = bconv[d];
  }
  __syncthreads();
  int idx = blockIdx.x * blockDim.x + tid;
  if (idx >= TBN_TOT) return;
  int t2 = idx / BN_TOT;
  int bn = idx - t2 * BN_TOT;
  int b = bn >> 9;
  int node2 = bn & 511;
  // torch raw .view(B, C*p, N, n): z[b,cp,node2,t2] maps through flat offset
  float src[9];
#pragma unroll
  for (int cp = 0; cp < 9; ++cp) {
    int O  = cp * (NNODES * T_STEPS) + node2 * T_STEPS + t2;     // N*n = 16896
    int c  = O / (NNODES * T_STEPS * 3);                          // 50688
    int r1 = O - c * (NNODES * T_STEPS * 3);
    int node = r1 / 99;                                           // n*p = 99
    int l = r1 - node * 99;                                       // padded pos
    if (l > 95) l = 95;                                           // replication pad
    src[cp] = x[(((size_t)b * CIN + c) * NNODES + node) * LSEQ + l];
  }
  __bf16* orow = xt + (size_t)idx * DMODEL;
  for (int d = 0; d < DMODEL; ++d) {
    float acc = sBc[d];
#pragma unroll
    for (int cp = 0; cp < 9; ++cp) acc += sWc[d * 9 + cp] * src[cp];
    orow[d] = (__bf16)(acc * sScale[d] + sShift[d]);
  }
}

// ---------------- kernel 2: projections k/v via WMMA + LN + softplus ----------------
// one WG (8 waves) per (t, b, 16-node block); wave w owns output cols [w*16, w*16+16)
__global__ void __launch_bounds__(256)
k2_proj(const __bf16* __restrict__ xt,
        const __bf16* __restrict__ Wkb, const float* __restrict__ bk,
        const __bf16* __restrict__ Wvb, const float* __restrict__ bv,
        const float* __restrict__ klng, const float* __restrict__ klnb,
        const float* __restrict__ vlng, const float* __restrict__ vlnb,
        __bf16* __restrict__ kout, __bf16* __restrict__ vout,
        float* __restrict__ vmax) {
  __shared__ alignas(16) __bf16 sX[16 * DMODEL];     // 4KB
  __shared__ float sOut[2][16][DMODEL];              // 16KB raw GEMM results
  __shared__ float redS[2][16][16], redQ[2][16][16]; // 4KB
  __shared__ float sMu[2][16], sRs[2][16];
  __shared__ float redM[16][16];

  int wg = blockIdx.x;
  int nb = wg & 31;
  int tb = wg >> 5;                                  // t*16 + b
  int nodebase = nb * 16;
  int tid = threadIdx.x;
  int lane = tid & 31, w = tid >> 5;

  const __bf16* Xblk = xt + ((size_t)tb * NNODES + nodebase) * DMODEL;
  for (int e = tid; e < 16 * DMODEL; e += 256) sX[e] = Xblk[e];
  __syncthreads();

  v16bf af[4];
#pragma unroll
  for (int ks = 0; ks < 4; ++ks) af[ks] = load_frag_rowmajor(sX, DMODEL, ks * 32, lane);

  const v8f zacc = {0.f, 0.f, 0.f, 0.f, 0.f, 0.f, 0.f, 0.f};
  const __bf16* Wmats[2] = {Wkb, Wvb};
  const float*  Bias[2]  = {bk, bv};
#pragma unroll
  for (int p = 0; p < 2; ++p) {
    v8f acc = zacc;
    const __bf16* Wb = Wmats[p] + (size_t)(w * 16) * DMODEL;  // B[k][n] = W[n][k]
#pragma unroll
    for (int ks = 0; ks < 4; ++ks) {
      v16bf bfr = load_frag_rowmajor(Wb, DMODEL, ks * 32, lane);
      acc = __builtin_amdgcn_wmma_f32_16x16x32_bf16(false, af[ks], false, bfr,
                                                    (short)0, acc, false, false);
    }
    int n  = w * 16 + (lane & 15);
    int hi = lane >> 4;
    float bias = Bias[p][n];
#pragma unroll
    for (int v = 0; v < 8; ++v) sOut[p][hi * 8 + v][n] = acc[v] + bias;
  }
  __syncthreads();

  // LayerNorm over 128 cols: 16 threads per row, 8 cols each
  int r = tid >> 4, seg = tid & 15;
#pragma unroll
  for (int p = 0; p < 2; ++p) {
    float s = 0.f, q = 0.f;
#pragma unroll
    for (int e = 0; e < 8; ++e) { float v = sOut[p][r][seg * 8 + e]; s += v; q += v * v; }
    redS[p][r][seg] = s; redQ[p][r][seg] = q;
  }
  __syncthreads();
  if (tid < 32) {
    int p = tid >> 4, rr = tid & 15;
    float s = 0.f, q = 0.f;
    for (int e = 0; e < 16; ++e) { s += redS[p][rr][e]; q += redQ[p][rr][e]; }
    float mu = s * (1.f / DMODEL);
    float var = q * (1.f / DMODEL) - mu * mu;
    sMu[p][rr] = mu; sRs[p][rr] = rsqrtf(var + EPS_F);
  }
  __syncthreads();

  size_t obase = ((size_t)tb * NNODES + nodebase + r) * DMODEL;
  float vmx = 0.f;
#pragma unroll
  for (int e = 0; e < 8; ++e) {
    int col = seg * 8 + e;
    float kr = (sOut[0][r][col] - sMu[0][r]) * sRs[0][r] * klng[col] + klnb[col];
    kout[obase + col] = (__bf16)(softplus_f(kr) * KSCALE);
    float vr = (sOut[1][r][col] - sMu[1][r]) * sRs[1][r] * vlng[col] + vlnb[col];
    float vv = softplus_f(vr);
    vout[obase + col] = (__bf16)vv;
    vmx = fmaxf(vmx, fabsf(vv));
  }
  redM[r][seg] = vmx;
  __syncthreads();
  if (tid < 16) {
    float m = 0.f;
    for (int e = 0; e < 16; ++e) m = fmaxf(m, redM[tid][e]);
    vmax[(size_t)tb * NNODES + nodebase + tid] = m;
  }
}

// ---------------- kernel 3a: input gates ----------------
__global__ void k3a_gate(const __bf16* __restrict__ xt, const float* __restrict__ Wi,
                         const float* __restrict__ bi, float* __restrict__ Ig) {
  __shared__ float sWi[DMODEL];
  int tid = threadIdx.x;
  if (tid < DMODEL) sWi[tid] = Wi[tid];
  __syncthreads();
  int idx = blockIdx.x * blockDim.x + tid;
  if (idx >= TBN_TOT) return;
  const __bf16* row = xt + (size_t)idx * DMODEL;
  float acc = bi[0];
  for (int d = 0; d < DMODEL; ++d) acc += (float)row[d] * sWi[d];
  Ig[idx] = 1.f / (1.f + expf(-acc));
}

// ---------------- kernel 3b: norm recurrence + suffix-product weights ----------------
__global__ void k3b_recur(const float* __restrict__ Ig, const float* __restrict__ vmax,
                          float* __restrict__ wgt, float* __restrict__ tmpg) {
  int bn = blockIdx.x * blockDim.x + threadIdx.x;
  if (bn >= BN_TOT) return;
  float Iv[T_STEPS];
  float norm = 0.f;
  for (int t = 0; t < T_STEPS; ++t) {
    int tbn = t * BN_TOT + bn;
    float I = Ig[tbn];
    Iv[t] = I;
    norm = (1.f - I) * norm + I * vmax[tbn];
    float tmp = norm - 1.f;
    tmpg[tbn] = (tmp > 0.f ? tmp : 0.f) + 1.f;
  }
  float prod = 1.f;
  for (int t = T_STEPS - 1; t >= 0; --t) {
    wgt[t * BN_TOT + bn] = Iv[t] * prod;
    prod *= (1.f - Iv[t]);
  }
}

// ---------------- kernel 4: now = relu(LN_N(k v^T)); threshold; cell accumulate ----
// one WG (8 waves) per (b, 16-row block). Cell slice (16x512) lives in VGPRs.
// v chunks are double-buffered in LDS via GLOBAL_LOAD_ASYNC_TO_LDS_B128 (ASYNCcnt).
__global__ void __launch_bounds__(256, 1)
k4_scan(const __bf16* __restrict__ kall, const __bf16* __restrict__ vall,
        const float* __restrict__ wgt, const float* __restrict__ tmpg,
        const float* __restrict__ nlng, const float* __restrict__ nlnb,
        float* __restrict__ out) {
  __shared__ alignas(16) __bf16 sK[16 * DMODEL];       // 4KB
  __shared__ alignas(16) __bf16 sV[2][128 * DMODEL];   // 2 x 32KB double buffer
  __shared__ float redS[16][8], redQ[16][8], redMn[16][8], redMx[16][8];
  __shared__ float rowMu[16], rowRs[16], rowW[16], rowTmp[16], rowThr[16];

  int b  = blockIdx.x >> 5;
  int rb = blockIdx.x & 31;
  int rowbase = rb * 16;
  int tid = threadIdx.x, lane = tid & 31, w = tid >> 5;
  int hi = lane >> 4, l15 = lane & 15;

  uint32_t svb0 = (uint32_t)(uintptr_t)&sV[0][0];
  uint32_t svb1 = (uint32_t)(uintptr_t)&sV[1][0];
  uint32_t skb  = (uint32_t)(uintptr_t)&sK[0];

  // LN gamma/beta are indexed by column j, fixed per (wave, chunk, lane)
  float g_c[4], bt_c[4];
#pragma unroll
  for (int c = 0; c < 4; ++c) {
    int col = c * 128 + w * 16 + l15;
    g_c[c]  = nlng[col];
    bt_c[c] = nlnb[col];
  }

  float cell[4][8];
#pragma unroll
  for (int c = 0; c < 4; ++c)
#pragma unroll
    for (int v = 0; v < 8; ++v) cell[c][v] = 0.f;

  const v8f zacc = {0.f, 0.f, 0.f, 0.f, 0.f, 0.f, 0.f, 0.f};

  for (int t = 0; t < T_STEPS; ++t) {
    __syncthreads();   // all waves done reading both sV buffers and sK from prev t
    const __bf16* Kblk  = kall + (((size_t)t * BATCH + b) * NNODES + rowbase) * DMODEL;
    const __bf16* Vbase = vall + ((size_t)t * BATCH + b) * NNODES * DMODEL;

    // stage sK (1 async op) + v chunks 0,1 (8 async ops each); per-wave asynccnt = 17
    {
      uint32_t l = skb + (uint32_t)tid * 16u;
      uint64_t g = (uint64_t)(uintptr_t)Kblk + (uint64_t)tid * 16u;
      asm volatile("global_load_async_to_lds_b128 %0, %1, off"
                   :: "v"(l), "v"(g) : "memory");
    }
    async_copy_chunk(Vbase + 0 * 128 * DMODEL, svb0, tid);
    async_copy_chunk(Vbase + 1 * 128 * DMODEL, svb1, tid);

    v8f raw[4];
    v16bf af[4];
#pragma unroll
    for (int c = 0; c < 4; ++c) raw[c] = zacc;

    auto compute_chunk = [&](int c, int buf) {
      const __bf16* Vt = &sV[buf][0] + (size_t)(w * 16) * DMODEL;  // B[k][j] = v[j][k]
#pragma unroll
      for (int ks = 0; ks < 4; ++ks) {
        v16bf bfr = load_frag_rowmajor(Vt, DMODEL, ks * 32, lane);
        raw[c] = __builtin_amdgcn_wmma_f32_16x16x32_bf16(false, af[ks], false, bfr,
                                                         (short)0, raw[c], false, false);
      }
    };

    // chunk 0: wait for sK + chunk0 (in-order completion leaves chunk1's 8 pending)
    wait_asynccnt_8();
    __syncthreads();
#pragma unroll
    for (int ks = 0; ks < 4; ++ks) af[ks] = load_frag_rowmajor(sK, DMODEL, ks * 32, lane);
    compute_chunk(0, 0);

    // chunk 1: all pending copies done; prefetch chunk2 into buf0 (overlaps compute)
    wait_asynccnt_0();
    __syncthreads();
    async_copy_chunk(Vbase + 2 * 128 * DMODEL, svb0, tid);
    compute_chunk(1, 1);

    // chunk 2: prefetch chunk3 into buf1 (overlaps compute)
    wait_asynccnt_0();
    __syncthreads();
    async_copy_chunk(Vbase + 3 * 128 * DMODEL, svb1, tid);
    compute_chunk(2, 0);

    // chunk 3
    wait_asynccnt_0();
    __syncthreads();
    compute_chunk(3, 1);

    // row sums / sumsq over N=512 (pre-LN)
    float ps[8], pq[8];
#pragma unroll
    for (int v = 0; v < 8; ++v) {
      float s = raw[0][v] + raw[1][v] + raw[2][v] + raw[3][v];
      float q = raw[0][v] * raw[0][v] + raw[1][v] * raw[1][v] +
                raw[2][v] * raw[2][v] + raw[3][v] * raw[3][v];
#pragma unroll
      for (int m = 1; m < 16; m <<= 1) {
        s += __shfl_xor(s, m, 32);
        q += __shfl_xor(q, m, 32);
      }
      ps[v] = s; pq[v] = q;
    }
    if (l15 == 0) {
#pragma unroll
      for (int v = 0; v < 8; ++v) { redS[hi * 8 + v][w] = ps[v]; redQ[hi * 8 + v][w] = pq[v]; }
    }
    __syncthreads();
    if (tid < 16) {
      float s = 0.f, q = 0.f;
#pragma unroll
      for (int e = 0; e < 8; ++e) { s += redS[tid][e]; q += redQ[tid][e]; }
      float mu = s * (1.f / NNODES);
      float var = q * (1.f / NNODES) - mu * mu;
      rowMu[tid] = mu;
      rowRs[tid] = rsqrtf(var + EPS_F);
      int tbn = t * BN_TOT + b * NNODES + rowbase + tid;
      rowW[tid]   = wgt[tbn];
      rowTmp[tid] = tmpg[tbn];
    }
    __syncthreads();

    // normalize + relu in registers; row min/max
    float pmn[8], pmx[8];
#pragma unroll
    for (int v = 0; v < 8; ++v) {
      int rr = hi * 8 + v;
      float mu = rowMu[rr], rs = rowRs[rr];
      float mn = 3.4e38f, mx = -3.4e38f;
#pragma unroll
      for (int c = 0; c < 4; ++c) {
        float val = (raw[c][v] - mu) * rs * g_c[c] + bt_c[c];
        val = val > 0.f ? val : 0.f;
        raw[c][v] = val;
        mn = fminf(mn, val); mx = fmaxf(mx, val);
      }
#pragma unroll
      for (int m = 1; m < 16; m <<= 1) {
        mn = fminf(mn, __shfl_xor(mn, m, 32));
        mx = fmaxf(mx, __shfl_xor(mx, m, 32));
      }
      pmn[v] = mn; pmx[v] = mx;
    }
    if (l15 == 0) {
#pragma unroll
      for (int v = 0; v < 8; ++v) { redMn[hi * 8 + v][w] = pmn[v]; redMx[hi * 8 + v][w] = pmx[v]; }
    }
    __syncthreads();
    if (tid < 16) {
      float mn = 3.4e38f, mx = -3.4e38f;
#pragma unroll
      for (int e = 0; e < 8; ++e) { mn = fminf(mn, redMn[tid][e]); mx = fmaxf(mx, redMx[tid][e]); }
      // nmm >= 0.5  <=>  now >= mn + 0.5*tmp*(mx-mn); degenerate row -> all zero (NaN in ref)
      rowThr[tid] = (mx > mn) ? (mn + 0.5f * rowTmp[tid] * (mx - mn)) : 3.4e38f;
    }
    __syncthreads();

#pragma unroll
    for (int v = 0; v < 8; ++v) {
      int rr = hi * 8 + v;
      float thr = rowThr[rr], wv = rowW[rr];
#pragma unroll
      for (int c = 0; c < 4; ++c)
        if (raw[c][v] >= thr) cell[c][v] += wv;
    }
  }

#pragma unroll
  for (int c = 0; c < 4; ++c) {
    int col = c * 128 + w * 16 + l15;
#pragma unroll
    for (int v = 0; v < 8; ++v) {
      int row = rowbase + hi * 8 + v;
      out[((size_t)b * NNODES + row) * NNODES + col] = cell[c][v];
    }
  }
}

// ---------------- launch ----------------
extern "C" void kernel_launch(void* const* d_in, const int* in_sizes, int n_in,
                              void* d_out, int out_size, void* d_ws, size_t ws_size,
                              hipStream_t stream) {
  const float* x     = (const float*)d_in[0];
  const float* Wk    = (const float*)d_in[1];
  const float* bk    = (const float*)d_in[2];
  const float* Wv    = (const float*)d_in[3];
  const float* bv    = (const float*)d_in[4];
  const float* Wi    = (const float*)d_in[5];
  const float* bi    = (const float*)d_in[6];
  const float* Wconv = (const float*)d_in[7];
  const float* bconv = (const float*)d_in[8];
  const float* bng   = (const float*)d_in[9];
  const float* bnb   = (const float*)d_in[10];
  const float* bnm   = (const float*)d_in[11];
  const float* bnv   = (const float*)d_in[12];
  const float* klng  = (const float*)d_in[13];
  const float* klnb  = (const float*)d_in[14];
  const float* vlng  = (const float*)d_in[15];
  const float* vlnb  = (const float*)d_in[16];
  const float* nlng  = (const float*)d_in[17];
  const float* nlnb  = (const float*)d_in[18];
  float* out = (float*)d_out;

  const size_t BF_MAT = (size_t)TBN_TOT * DMODEL * sizeof(__bf16);  // 69.2MB
  const size_t F_TBN  = (size_t)TBN_TOT * sizeof(float);            // 1.08MB

  char* ws = (char*)d_ws;
  size_t off = 0;
  __bf16* xt   = (__bf16*)(ws + off); off += BF_MAT;
  __bf16* kall = (__bf16*)(ws + off); off += BF_MAT;
  __bf16* vall = (__bf16*)(ws + off); off += BF_MAT;
  __bf16* Wkb  = (__bf16*)(ws + off); off += DMODEL * DMODEL * sizeof(__bf16);
  __bf16* Wvb  = (__bf16*)(ws + off); off += DMODEL * DMODEL * sizeof(__bf16);
  float*  Ig   = (float*)(ws + off);  off += F_TBN;
  float*  vmax = (float*)(ws + off);  off += F_TBN;
  float*  wgt  = (float*)(ws + off);  off += F_TBN;
  float*  tmpg = (float*)(ws + off);  off += F_TBN;

  k0_convert<<<64, 256, 0, stream>>>(Wk, Wv, Wkb, Wvb);
  k1_patch_conv_bn<<<(TBN_TOT + 255) / 256, 256, 0, stream>>>(
      x, Wconv, bconv, bng, bnb, bnm, bnv, xt);
  k2_proj<<<T_STEPS * BATCH * 32, 256, 0, stream>>>(
      xt, Wkb, bk, Wvb, bv, klng, klnb, vlng, vlnb, kall, vall, vmax);
  k3a_gate<<<(TBN_TOT + 255) / 256, 256, 0, stream>>>(xt, Wi, bi, Ig);
  k3b_recur<<<(BN_TOT + 255) / 256, 256, 0, stream>>>(Ig, vmax, wgt, tmpg);
  k4_scan<<<BATCH * 32, 256, 0, stream>>>(kall, vall, wgt, tmpg, nlng, nlnb, out);
}
